// CustomConv2d_84018150244418
// MI455X (gfx1250) — matrix operand order repaired and checked by reading, not compiled, not run on hardware
//
#include <hip/hip_runtime.h>

typedef __attribute__((ext_vector_type(16))) __bf16 v16bf;
typedef __attribute__((ext_vector_type(8)))  float  v8f;

#define CIN   128
#define COUT  256
#define HW    128
#define ASTRIDE 40            // LDS row stride (bf16): 80B rows, 16B aligned, conflict-free
#define BUFE  (130 * ASTRIDE) // elements per plane per buffer (pixels -1..128 at idx+1)
#define TSTRIDE 136           // prep_x LDS tile row stride

__device__ __forceinline__ void split_bf16(float x, __bf16& hi, __bf16& lo) {
    hi = (__bf16)x;
    lo = (__bf16)(x - (float)hi);
}

// ---------------- weight prep: fold [0.7,1,0.7]^2, split fp32->bf16 hi/lo, OIHW -> [p][cout][cin]
__global__ void prep_weights(const float* __restrict__ w,
                             __bf16* __restrict__ whi, __bf16* __restrict__ wlo) {
    int idx = blockIdx.x * blockDim.x + threadIdx.x;
    if (idx >= COUT * CIN * 9) return;
    int j = idx % 3;
    int i = (idx / 3) % 3;
    int c = (idx / 9) % CIN;
    int o = idx / (9 * CIN);
    float ki = (i == 1) ? 1.0f : 0.7f;
    float kj = (j == 1) ? 1.0f : 0.7f;
    float v = w[idx] * ki * kj;
    __bf16 h, l;
    split_bf16(v, h, l);
    int d = ((i * 3 + j) * COUT + o) * CIN + c;
    whi[d] = h;
    wlo[d] = l;
}

// ---------------- input prep: NCHW fp32 -> NHWC bf16 hi/lo (LDS transpose, coalesced R+W)
__global__ __launch_bounds__(256)
void prep_x(const float* __restrict__ x, __bf16* __restrict__ xhi, __bf16* __restrict__ xlo) {
    __shared__ __bf16 tHi[64 * TSTRIDE];
    __shared__ __bf16 tLo[64 * TSTRIDE];
    const int wb  = blockIdx.x;
    const int h   = blockIdx.y;
    const int n   = blockIdx.z;
    const int tid = threadIdx.x;
    #pragma unroll
    for (int rep = 0; rep < 32; ++rep) {
        int lin = tid + rep * 256;               // [c 128][w 64]
        int c   = lin >> 6;
        int wl  = lin & 63;
        float v = x[(((size_t)n * CIN + c) * HW + h) * HW + wb * 64 + wl];
        __bf16 hb, lb;
        split_bf16(v, hb, lb);
        tHi[wl * TSTRIDE + c] = hb;
        tLo[wl * TSTRIDE + c] = lb;
    }
    __syncthreads();
    #pragma unroll
    for (int rep = 0; rep < 4; ++rep) {
        int lin = tid + rep * 256;               // [w 64][c/8 16]
        int wl  = lin >> 4;
        int c8  = (lin & 15) * 8;
        uint4 vh = *(uint4*)&tHi[wl * TSTRIDE + c8];
        uint4 vl = *(uint4*)&tLo[wl * TSTRIDE + c8];
        size_t d = (((size_t)n * HW + h) * HW + wb * 64 + wl) * CIN + c8;
        *(uint4*)&xhi[d] = vh;
        *(uint4*)&xlo[d] = vl;
    }
}

union Frag { uint4 u[2]; v16bf v; };

// ---------------- main conv: implicit GEMM, bf16x3 split, double-buffered async LDS staging
__global__ __launch_bounds__(256)
void conv3x3_wmma(const __bf16* __restrict__ xhi, const __bf16* __restrict__ xlo,
                  const __bf16* __restrict__ whi, const __bf16* __restrict__ wlo,
                  const float* __restrict__ bias, float* __restrict__ out) {
    __shared__ __bf16 aHi[2 * BUFE];
    __shared__ __bf16 aLo[2 * BUFE];

    const int tid   = threadIdx.x;
    const int lane  = tid & 31;
    const int wv    = tid >> 5;
    const int waveM = wv & 3;
    const int waveN = wv >> 2;
    const int oblk  = blockIdx.x;
    const int h     = blockIdx.y;
    const int nimg  = blockIdx.z;
    const int lh    = lane >> 4;
    const int ln    = lane & 15;

    // per-lane constant B base (bytes); all loop loads use compile-time immediates off this
    const int   obase = oblk * 128 + waveN * 64 + ln;
    const char* wHb = (const char*)whi + ((size_t)obase * CIN + lh * 16) * 2;
    const char* wLb = (const char*)wlo + ((size_t)obase * CIN + lh * 16) * 2;
    // per-lane constant A base (elements into shared planes)
    const int albase = (waveM * 32 + ln) * ASTRIDE + lh * 16;

    // bias folded into accumulator init (C operand of first WMMA)
    v8f acc[2][4];
    #pragma unroll
    for (int ni = 0; ni < 4; ++ni) {
        float bv = bias[obase + ni * 16];
        v8f a0;
        #pragma unroll
        for (int r = 0; r < 8; ++r) a0[r] = bv;
        acc[0][ni] = a0;
        acc[1][ni] = a0;
    }

    // halo rows (pixel idx 0 and 129) of both buffers stay zero forever
    if (tid < 32) {
        uint4 z = {0u,0u,0u,0u};
        int pl = tid >> 4;
        int bf = (tid >> 3) & 1;
        int mp = ((tid >> 2) & 1) ? 129 : 0;
        int q  = tid & 3;
        __bf16* l = (pl ? aLo : aHi) + bf * BUFE + mp * ASTRIDE + q * 8;
        *(uint4*)l = z;
    }

    auto stage = [&](int s) {
        const int dy = s >> 2, cb = s & 3, b = s & 1;
        const int h2 = h + dy - 1;
        if ((unsigned)h2 < (unsigned)HW) {
            #pragma unroll
            for (int it = 0; it < 4; ++it) {
                int ii = tid + it * 256;                 // 1024 x 16B chunks
                int pl = ii >> 9;
                int mp = 1 + ((ii >> 2) & 127);
                int q  = ii & 3;
                const __bf16* g = (pl ? xlo : xhi) +
                    (((size_t)nimg * HW + h2) * HW + (mp - 1)) * CIN + cb * 32 + q * 8;
                __bf16* l = (pl ? aLo : aHi) + b * BUFE + mp * ASTRIDE + q * 8;
                unsigned          loff = (unsigned)(unsigned long long)l;
                unsigned long long ga  = (unsigned long long)g;
                asm volatile("global_load_async_to_lds_b128 %0, %1, off"
                             :: "v"(loff), "v"(ga) : "memory");
            }
        } else {
            uint4 z = {0u,0u,0u,0u};
            #pragma unroll
            for (int it = 0; it < 4; ++it) {
                int ii = tid + it * 256;
                int pl = ii >> 9;
                int mp = 1 + ((ii >> 2) & 127);
                int q  = ii & 3;
                __bf16* l = (pl ? aLo : aHi) + b * BUFE + mp * ASTRIDE + q * 8;
                *(uint4*)l = z;
            }
        }
    };

    stage(0);
    asm volatile("s_wait_asynccnt 0x0" ::: "memory");
    __syncthreads();

    for (int s = 0; s < 12; ++s) {
        if (s < 11) stage(s + 1);                 // prefetch next slab into other buffer

        const int dy = s >> 2, cb = s & 3, b = s & 1;
        const char*   cwh = wHb + (size_t)dy * (3 * COUT * CIN * 2) + cb * 64;
        const char*   cwl = wLb + (size_t)dy * (3 * COUT * CIN * 2) + cb * 64;
        const __bf16* aH  = aHi + albase + b * BUFE;
        const __bf16* aL  = aLo + albase + b * BUFE;

        #pragma unroll
        for (int dx = 0; dx < 3; ++dx) {
            Frag bh[4], bl[4], ah[2], al[2];
            #pragma unroll
            for (int ni = 0; ni < 4; ++ni) {      // all immediates: dx*65536 + ni*4096 (+16)
                bh[ni].u[0] = *(const uint4*)(cwh + dx * (COUT * CIN * 2) + ni * (16 * CIN * 2));
                bh[ni].u[1] = *(const uint4*)(cwh + dx * (COUT * CIN * 2) + ni * (16 * CIN * 2) + 16);
                bl[ni].u[0] = *(const uint4*)(cwl + dx * (COUT * CIN * 2) + ni * (16 * CIN * 2));
                bl[ni].u[1] = *(const uint4*)(cwl + dx * (COUT * CIN * 2) + ni * (16 * CIN * 2) + 16);
            }
            #pragma unroll
            for (int mi = 0; mi < 2; ++mi) {      // ds offsets immediate: (mi*16+dx)*80 (+16)
                ah[mi].u[0] = *(const uint4*)(aH + (mi * 16 + dx) * ASTRIDE);
                ah[mi].u[1] = *(const uint4*)(aH + (mi * 16 + dx) * ASTRIDE + 8);
                al[mi].u[0] = *(const uint4*)(aL + (mi * 16 + dx) * ASTRIDE);
                al[mi].u[1] = *(const uint4*)(aL + (mi * 16 + dx) * ASTRIDE + 8);
            }
            #pragma unroll
            for (int mi = 0; mi < 2; ++mi)
                #pragma unroll
                for (int ni = 0; ni < 4; ++ni) {
                    acc[mi][ni] = __builtin_amdgcn_wmma_f32_16x16x32_bf16(
                        false, ah[mi].v, false, bh[ni].v, (short)0, acc[mi][ni], false, false);
                    acc[mi][ni] = __builtin_amdgcn_wmma_f32_16x16x32_bf16(
                        false, ah[mi].v, false, bl[ni].v, (short)0, acc[mi][ni], false, false);
                    acc[mi][ni] = __builtin_amdgcn_wmma_f32_16x16x32_bf16(
                        false, al[mi].v, false, bh[ni].v, (short)0, acc[mi][ni], false, false);
                }
        }
        asm volatile("s_wait_asynccnt 0x0" ::: "memory");
        __syncthreads();
    }

    // epilogue: pure stores (bias already accumulated); 8 consecutive pixels per lane
    #pragma unroll
    for (int mi = 0; mi < 2; ++mi) {
        const int mrow = waveM * 32 + mi * 16 + lh * 8;
        #pragma unroll
        for (int ni = 0; ni < 4; ++ni) {
            const int o = obase + ni * 16;
            float* po = out + ((((size_t)nimg * COUT + o) * HW + h) * HW + mrow);
            float4 f0, f1;
            f0.x = acc[mi][ni][0]; f0.y = acc[mi][ni][1];
            f0.z = acc[mi][ni][2]; f0.w = acc[mi][ni][3];
            f1.x = acc[mi][ni][4]; f1.y = acc[mi][ni][5];
            f1.z = acc[mi][ni][6]; f1.w = acc[mi][ni][7];
            *(float4*)(po)     = f0;
            *(float4*)(po + 4) = f1;
        }
    }
}

extern "C" void kernel_launch(void* const* d_in, const int* in_sizes, int n_in,
                              void* d_out, int out_size, void* d_ws, size_t ws_size,
                              hipStream_t stream) {
    (void)in_sizes; (void)n_in; (void)out_size; (void)ws_size;
    const float* x    = (const float*)d_in[0];
    const float* w    = (const float*)d_in[1];
    const float* bias = (const float*)d_in[2];
    float* out = (float*)d_out;

    const size_t wn = (size_t)9 * COUT * CIN;
    const size_t xn = (size_t)16 * HW * HW * CIN;
    char* ws = (char*)d_ws;
    __bf16* whi = (__bf16*)(ws);
    __bf16* wlo = (__bf16*)(ws + wn * 2);
    __bf16* xhi = (__bf16*)(ws + wn * 4);
    __bf16* xlo = (__bf16*)(ws + wn * 4 + xn * 2);

    prep_weights<<<(int)((wn + 255) / 256), 256, 0, stream>>>(w, whi, wlo);
    prep_x<<<dim3(2, HW, 16), 256, 0, stream>>>(x, xhi, xlo);
    conv3x3_wmma<<<dim3(COUT / 128, HW, 16), 256, 0, stream>>>(xhi, xlo, whi, wlo, bias, out);
}
